// CTBlock_34265249087836
// MI455X (gfx1250) — matrix operand order repaired and checked by reading, not compiled, not run on hardware
//
#include <hip/hip_runtime.h>
#include <hip/hip_bf16.h>
#include <math.h>

// ---------------- problem constants ----------------
#define Bq   256
#define Cc   248
#define Pp   50
#define Tt   4
#define Hh   8
#define Dd   31
#define DHd  124
#define PT   200          // Pp*Tt
#define NPT  51200        // Bq*PT
#define C3   744          // 3*Cc
#define FFD  16
#define KC   1240         // Cc*5 (fused conv K)
#define KCP  1248         // padded to 32
#define TOT  12697600     // Bq*Cc*PT
#define TOTY 3276800      // 4*Bq*FFD*PT
#define WROWS 768         // padded rows of f16 qkv weight
#define CPAD 256          // padded channels for im2col buffer
#define TPAD 8            // padded t-extent (halo 2 each side)

typedef __attribute__((ext_vector_type(16))) _Float16 v16h;
typedef __attribute__((ext_vector_type(8)))  float    v8f;

__device__ __forceinline__ v8f wmma32(v16h a, v16h b, v8f c) {
  // D = A(16x32 f16) * B(32x16 f16) + C(16x16 f32)
  return __builtin_amdgcn_wmma_f32_16x16x32_f16(false, a, false, b, (short)0, c,
                                                false, false);
}

// A-fragment: tile stored [m][k] row-major in LDS. Lane L -> row m=L&15,
// K = (L>=16?16:0) + j contiguous.
__device__ __forceinline__ v16h frag_A(const _Float16* t, int ld, int m0, int k0, int lane) {
  int m  = lane & 15;
  int kb = (lane >> 4) << 4;
  const _Float16* p = t + (size_t)(m0 + m) * ld + k0 + kb;
  v16h f;
#pragma unroll
  for (int j = 0; j < 16; ++j) f[j] = p[j];
  return f;
}

// B-fragment: tile stored [n][k] row-major (i.e. B^T). Lane L -> col n=L&15,
// K = (L>=16?16:0) + j contiguous.
__device__ __forceinline__ v16h frag_B(const _Float16* t, int ld, int n0, int k0, int lane) {
  int n  = lane & 15;
  int kb = (lane >> 4) << 4;
  const _Float16* p = t + (size_t)(n0 + n) * ld + k0 + kb;
  v16h f;
#pragma unroll
  for (int j = 0; j < 16; ++j) f[j] = p[j];
  return f;
}

// ---------------- K0: weight prep ----------------
// whp : padded [768][256] f16 qkv weight (zero outside 744x248)
// wcc : fused conv1(3-tap, embedded)+conv2(5-tap) weight [16][1248] f16, zero-padded
// relp: rel-emb padded [112][128] f16
__global__ __launch_bounds__(256) void prep_kernel(const float* __restrict__ wqkv,
                                                   const float* __restrict__ wc1,
                                                   const float* __restrict__ wc2,
                                                   const float* __restrict__ rel,
                                                   _Float16* __restrict__ whp,
                                                   _Float16* __restrict__ wcc,
                                                   _Float16* __restrict__ relp) {
  int i = blockIdx.x * 256 + threadIdx.x;
  if (i < WROWS * 256) {
    int o = i >> 8, k = i & 255;
    whp[i] = (o < C3 && k < Cc) ? (_Float16)wqkv[o * Cc + k] : (_Float16)0.f;
    return;
  }
  i -= WROWS * 256;
  if (i < FFD * KCP) {
    int f = i / KCP, k = i - f * KCP;
    float v = 0.f;
    if (k < KC) {
      int c = k / 5, tap = k - c * 5;
      if (f < 8) {                    // conv1: 3-tap embedded in 5-tap (pad 1 -> pad 2)
        if (tap >= 1 && tap <= 3) v = wc1[(f * Cc + c) * 3 + (tap - 1)];
      } else {                        // conv2: native 5-tap
        v = wc2[((f - 8) * Cc + c) * 5 + tap];
      }
    }
    wcc[i] = (_Float16)v;
    return;
  }
  i -= FFD * KCP;
  if (i < 112 * 128) {
    int r = i >> 7, dh = i & 127;
    relp[i] = (r < (2 * Pp - 1) && dh < DHd) ? (_Float16)rel[r * DHd + dh] : (_Float16)0.f;
  }
}

// ---------------- zero-fill (uint4 granularity) ----------------
__global__ __launch_bounds__(256) void fill_zero(uint4* __restrict__ p) {
  p[(size_t)blockIdx.x * 256 + threadIdx.x] = make_uint4(0u, 0u, 0u, 0u);
}

// ---------------- K1: QKV GEMM per batch: W[744x248] x X_b[248x200] ----------------
// Output directly in attention-native layout qkvT[b][s][h][p][dh] (dh = d*4+t, contiguous)
__global__ __launch_bounds__(256) void qkv_gemm(const float* __restrict__ x,
                                                const _Float16* __restrict__ whp,
                                                const float* __restrict__ bq,
                                                _Float16* __restrict__ qkvT) {
  __shared__ _Float16 As[64 * 32];   // [m][k]
  __shared__ _Float16 Bs[64 * 34];   // [n][k] padded
  int bm = blockIdx.x * 64;          // output-channel base
  int bn = blockIdx.y * 64;          // (p,t) base
  int b  = blockIdx.z;
  int tid  = threadIdx.x;
  int wave = tid >> 5, lane = tid & 31;
  int wr = wave >> 1, wc = wave & 1; // 4x2 wave grid -> each wave 16x32 of output
  const float* xb = x + (size_t)b * Cc * PT;
  v8f acc0 = {}, acc1 = {};
  for (int k0 = 0; k0 < 256; k0 += 32) {
    // A tile: unconditional vector copies from padded weight
#pragma unroll
    for (int it = 0; it < 2; ++it) {
      int i = tid + it * 256;
      int m = i >> 3, cj = (i & 7) * 4;
      *(uint2*)(As + m * 32 + cj) =
          *(const uint2*)(whp + (size_t)(bm + m) * 256 + k0 + cj);
    }
    // B tile: x_b[c][pt] -> Bs[n][k] (fp32 -> f16)
#pragma unroll
    for (int it = 0; it < 8; ++it) {
      int i = tid + it * 256;
      int n = i & 63, kk = i >> 6;
      int k = k0 + kk, pt = bn + n;
      Bs[n * 34 + kk] = (k < Cc && pt < PT) ? (_Float16)xb[k * PT + pt] : (_Float16)0.f;
    }
    if (k0 + 32 < 256) {
      __builtin_prefetch(xb + (k0 + 32) * PT + bn + (tid & 63), 0, 1);
      __builtin_prefetch(whp + (size_t)(bm + (tid >> 2)) * 256 + k0 + 32, 0, 1);
    }
    __syncthreads();
    v16h a  = frag_A(As, 32, wr * 16, 0, lane);
    v16h b0 = frag_B(Bs, 34, wc * 32, 0, lane);
    v16h b1 = frag_B(Bs, 34, wc * 32 + 16, 0, lane);
    acc0 = wmma32(a, b0, acc0);
    acc1 = wmma32(a, b1, acc1);
    __syncthreads();
  }
  int mm = (lane >> 4) << 3, nn = lane & 15;
#pragma unroll
  for (int r = 0; r < 8; ++r) {
    int o = bm + wr * 16 + r + mm;
    if (o < C3) {
      float bias = bq[o];
      int s  = o / Cc;
      int r2 = o - s * Cc;
      int hh = r2 / Dd;
      int d  = r2 - hh * Dd;
      size_t plane = (((size_t)b * 3 + s) * Hh + hh) * (Pp * DHd);
      int pt0 = bn + wc * 32 + nn;
      int pt1 = pt0 + 16;
      if (pt0 < PT)
        qkvT[plane + (size_t)(pt0 >> 2) * DHd + d * 4 + (pt0 & 3)] =
            (_Float16)(acc0[r] + bias);
      if (pt1 < PT)
        qkvT[plane + (size_t)(pt1 >> 2) * DHd + d * 4 + (pt1 & 3)] =
            (_Float16)(acc1[r] + bias);
    }
  }
}

// ---------------- K2: attention per (b,h) workgroup, fused residual ----------------
__global__ __launch_bounds__(256) void attn_kernel(const _Float16* __restrict__ qkvT,
                                                   const _Float16* __restrict__ relp,
                                                   const float* __restrict__ x,
                                                   float* __restrict__ pre1) {
  extern __shared__ char smem[];
  _Float16* qs  = (_Float16*)smem;        // 64x128 [p][dh]
  _Float16* ks  = qs + 64 * 128;          // 64x128 [p'][dh]
  _Float16* vs  = ks + 64 * 128;          // 64x128 [p'][dh]
  _Float16* at  = vs + 64 * 128;          // attn 64x64 [p][p']
  float*    Sd  = (float*)(at + 64 * 64); // dots 64x64
  float*    Sb  = Sd + 64 * 64;           // rel bias 64x112
  int h = blockIdx.x, b = blockIdx.y;
  int tid  = threadIdx.x;
  int wave = tid >> 5, lane = tid & 31;
  const _Float16* q0 = qkvT + (((size_t)b * 3 + 0) * Hh + h) * (Pp * DHd);
  const _Float16* k0p = qkvT + (((size_t)b * 3 + 1) * Hh + h) * (Pp * DHd);
  const _Float16* v0 = qkvT + (((size_t)b * 3 + 2) * Hh + h) * (Pp * DHd);
  // vectorized fill: rows contiguous in dh, 4 halves (8B) per chunk
#pragma unroll
  for (int it = 0; it < 8; ++it) {
    int i = tid + it * 256;
    int row = i >> 5, j = i & 31;               // 32 chunks of 4 halves per row
    bool ok = (row < Pp) && (j < 31);           // 31*4 = 124 valid halves
    uint2 zq = make_uint2(0u, 0u), zk = zq, zv = zq;
    if (ok) {
      size_t o = (size_t)row * DHd + j * 4;
      zq = *(const uint2*)(q0 + o);
      zk = *(const uint2*)(k0p + o);
      zv = *(const uint2*)(v0 + o);
    }
    int d = row * 128 + j * 4;
    *(uint2*)(qs + d) = zq;
    *(uint2*)(ks + d) = zk;
    *(uint2*)(vs + d) = zv;
  }
  __syncthreads();
  // dots = q @ k^T : 4x4 tiles, 2 per wave
  for (int ii = 0; ii < 2; ++ii) {
    int idx = wave + ii * 8;
    int tr = idx >> 2, tc = idx & 3;
    v8f acc = {};
#pragma unroll
    for (int kk = 0; kk < 128; kk += 32) {
      v16h a  = frag_A(qs, 128, tr * 16, kk, lane);
      v16h bb = frag_B(ks, 128, tc * 16, kk, lane);
      acc = wmma32(a, bb, acc);
    }
    int mm = (lane >> 4) << 3, nn = lane & 15;
#pragma unroll
    for (int r = 0; r < 8; ++r)
      Sd[(tr * 16 + r + mm) * 64 + tc * 16 + nn] = acc[r];
  }
  // rel bias = q @ rel^T : 4x7 tiles
  for (int idx = wave; idx < 28; idx += 8) {
    int tr = idx / 7, tc = idx - tr * 7;
    v8f acc = {};
#pragma unroll
    for (int kk = 0; kk < 128; kk += 32) {
      v16h a  = frag_A(qs, 128, tr * 16, kk, lane);
      v16h bb = frag_B(relp, 128, tc * 16, kk, lane);
      acc = wmma32(a, bb, acc);
    }
    int mm = (lane >> 4) << 3, nn = lane & 15;
#pragma unroll
    for (int r = 0; r < 8; ++r)
      Sb[(tr * 16 + r + mm) * 112 + tc * 16 + nn] = acc[r];
  }
  __syncthreads();
  // softmax over p' (rel_to_abs: bias[p][p'] = Sb[p][p'-p+49]); scale AFTER bias
  const float SCALE = 0.17960530202677491f;  // 31^-0.5
  if (tid < 64) {
    int p = tid;
    if (p < Pp) {
      float mx = -1e30f;
      for (int j = 0; j < Pp; ++j) {
        float s = (Sd[p * 64 + j] + Sb[p * 112 + (j - p + Pp - 1)]) * SCALE;
        mx = fmaxf(mx, s);
      }
      float sum = 0.f;
      for (int j = 0; j < Pp; ++j) {
        float s = (Sd[p * 64 + j] + Sb[p * 112 + (j - p + Pp - 1)]) * SCALE;
        sum += __expf(s - mx);
      }
      float inv = 1.f / sum;
      for (int j = 0; j < 64; ++j) {
        float a = 0.f;
        if (j < Pp) {
          float s = (Sd[p * 64 + j] + Sb[p * 112 + (j - p + Pp - 1)]) * SCALE;
          a = __expf(s - mx) * inv;
        }
        at[p * 64 + j] = (_Float16)a;
      }
    } else {
      for (int j = 0; j < 64; ++j) at[p * 64 + j] = (_Float16)0.f;
    }
  }
  __syncthreads();
  // out = attn @ v : 4x8 tiles, fused residual store into pre-BN1 buffer
  for (int ii = 0; ii < 4; ++ii) {
    int idx = wave + ii * 8;
    int tr = idx >> 3, tc = idx & 7;
    v8f acc = {};
#pragma unroll
    for (int kk = 0; kk < 64; kk += 32) {
      v16h a  = frag_A(at, 64, tr * 16, kk, lane);
      v16h bb = frag_B(vs, 128, tc * 16, kk, lane);
      acc = wmma32(a, bb, acc);
    }
    int mm = (lane >> 4) << 3, nn = lane & 15;
#pragma unroll
    for (int r = 0; r < 8; ++r) {
      int p = tr * 16 + r + mm;
      int dh = tc * 16 + nn;
      if (p < Pp && dh < DHd) {
        size_t o = ((size_t)b * Cc + h * Dd + (dh >> 2)) * PT + p * 4 + (dh & 3);
        pre1[o] = acc[r] + x[o];
      }
    }
  }
}

// ---------------- BN stats: one block per channel -> scale/shift ----------------
__global__ __launch_bounds__(256) void bn_stats(const float* __restrict__ in,
                                                const float* __restrict__ g,
                                                const float* __restrict__ be,
                                                float* __restrict__ sc,
                                                float* __restrict__ sh,
                                                int C, int spatial, int Bn) {
  __shared__ float rs[256], rq[256];
  int c = blockIdx.x, tid = threadIdx.x;
  float s = 0.f, ss = 0.f;
  int total = Bn * spatial;
  for (int i = tid; i < total; i += 256) {
    int b = i / spatial, j = i - b * spatial;
    float v = in[((size_t)b * C + c) * spatial + j];
    s += v; ss += v * v;
  }
  rs[tid] = s; rq[tid] = ss;
  __syncthreads();
  for (int o = 128; o > 0; o >>= 1) {
    if (tid < o) { rs[tid] += rs[tid + o]; rq[tid] += rq[tid + o]; }
    __syncthreads();
  }
  if (tid == 0) {
    float n = (float)total;
    float m = rs[0] / n;
    float var = rq[0] / n - m * m;
    float k = g[c] * rsqrtf(var + 1e-5f);
    sc[c] = k;
    sh[c] = be[c] - m * k;
  }
}

// ---------------- BN1 apply -> x1 (f32) + halo-padded f16 im2col buffer ----------------
// x1p layout: [b][CPAD=256][Pp][TPAD=8], valid data at t-slot 2..5, rest pre-zeroed
__global__ __launch_bounds__(256) void bn_apply_x1(const float* __restrict__ in,
                                                   const float* __restrict__ sc,
                                                   const float* __restrict__ sh,
                                                   float* __restrict__ x1,
                                                   _Float16* __restrict__ x1p) {
  size_t i = (size_t)blockIdx.x * 256 + threadIdx.x;
  int t = (int)(i & 3);
  size_t pc = i >> 2;
  int p = (int)(pc % Pp);
  size_t bc = pc / Pp;
  int c = (int)(bc % Cc);
  int b = (int)(bc / Cc);
  float v = in[i] * sc[c] + sh[c];
  x1[i] = v;
  x1p[(((size_t)b * CPAD + c) * Pp + p) * TPAD + 2 + t] = (_Float16)v;
}

// ---------------- K5: fused conv1(1x3)+conv2(1x5) as one im2col WMMA GEMM ----------------
// M=16 (8+8 out channels), K=1240 (c*5+tap), N=51200. Halo padding makes all
// im2col loads unconditional: slot = t + tap in [0,8).
__global__ __launch_bounds__(256) void cfe_gemm(const _Float16* __restrict__ x1p,
                                                const _Float16* __restrict__ wcc,
                                                const float* __restrict__ bc1,
                                                const float* __restrict__ bc2,
                                                float* __restrict__ y0) {
  __shared__ _Float16 As[16 * 32];    // [f][k]
  __shared__ _Float16 Bs[128 * 34];   // [n][k]
  int bn = blockIdx.x * 128;
  int tid = threadIdx.x;
  int wave = tid >> 5, lane = tid & 31;
  v8f acc = {};
  for (int kk0 = 0; kk0 < KCP; kk0 += 32) {
    if (tid < 128) {
      int f = tid >> 3, cj = (tid & 7) * 4;
      *(uint2*)(As + f * 32 + cj) = *(const uint2*)(wcc + f * KCP + kk0 + cj);
    }
#pragma unroll
    for (int it = 0; it < 16; ++it) {
      int i = tid + it * 256;
      int n = i & 127, kk = i >> 7;
      int k = kk0 + kk;                  // < 1248; c <= 249 < CPAD
      int c = k / 5, tap = k - c * 5;
      int g = bn + n;                    // global (b,p,t), t fastest
      int t = g & 3;
      int p = (g >> 2) % Pp;
      int bb = g / PT;
      Bs[n * 34 + kk] =
          x1p[(((size_t)bb * CPAD + c) * Pp + p) * TPAD + t + tap];
    }
    __syncthreads();
    v16h a  = frag_A(As, 32, 0, 0, lane);
    v16h bb = frag_B(Bs, 34, wave * 16, 0, lane);
    acc = wmma32(a, bb, acc);
    __syncthreads();
  }
  int mm = (lane >> 4) << 3, nn = lane & 15;
#pragma unroll
  for (int r = 0; r < 8; ++r) {
    int f = r + mm;                     // 0..15
    int n = bn + wave * 16 + nn;        // < 51200 exactly
    float bias = (f < 8) ? bc1[f] : bc2[f - 8];
    int t = n & 3, p = (n >> 2) % Pp, bb = n / PT;
    y0[((size_t)bb * FFD + f) * PT + p * 4 + t] = acc[r] + bias;
  }
}

// ---------------- BN_cfe apply + ELU ----------------
__global__ __launch_bounds__(256) void bn_elu(const float* __restrict__ y0,
                                              const float* __restrict__ sc,
                                              const float* __restrict__ sh,
                                              float* __restrict__ y) {
  size_t i = (size_t)blockIdx.x * 256 + threadIdx.x;
  int c = (int)((i / PT) % FFD);
  float v = y0[i] * sc[c] + sh[c];
  y[i] = v > 0.f ? v : expm1f(v);
}

// ---------------- conv3 (1x1, 16->248) + residual x1 ----------------
__global__ __launch_bounds__(256) void conv3_res(const float* __restrict__ y,
                                                 const float* __restrict__ w3,
                                                 const float* __restrict__ bc3,
                                                 const float* __restrict__ x1,
                                                 float* __restrict__ pre2) {
  __shared__ float ys[FFD * PT];
  int b = blockIdx.y;
  int c0 = blockIdx.x * 62;
  int tid = threadIdx.x;
  for (int i = tid; i < FFD * PT; i += 256) ys[i] = y[(size_t)b * FFD * PT + i];
  __syncthreads();
  for (int idx = tid; idx < 62 * PT; idx += 256) {
    int c = c0 + idx / PT;
    int j = idx % PT;
    float acc = bc3[c];
#pragma unroll
    for (int f = 0; f < FFD; ++f) acc += w3[c * FFD + f] * ys[f * PT + j];
    size_t o = ((size_t)b * Cc + c) * PT + j;
    pre2[o] = acc + x1[o];
  }
}

// ---------------- final BN2 apply -> d_out ----------------
__global__ __launch_bounds__(256) void bn_apply_out(const float* __restrict__ in,
                                                    const float* __restrict__ sc,
                                                    const float* __restrict__ sh,
                                                    float* __restrict__ out) {
  size_t i = (size_t)blockIdx.x * 256 + threadIdx.x;
  int c = (int)((i / PT) % Cc);
  out[i] = in[i] * sc[c] + sh[c];
}

extern "C" void kernel_launch(void* const* d_in, const int* in_sizes, int n_in,
                              void* d_out, int out_size, void* d_ws, size_t ws_size,
                              hipStream_t stream) {
  (void)in_sizes; (void)n_in; (void)out_size; (void)ws_size;
  const float* x        = (const float*)d_in[0];
  const float* w_qkv    = (const float*)d_in[1];
  const float* b_qkv    = (const float*)d_in[2];
  const float* rel_emb  = (const float*)d_in[3];
  const float* g1       = (const float*)d_in[4];
  const float* beta1    = (const float*)d_in[5];
  const float* w_c1     = (const float*)d_in[6];
  const float* bc1      = (const float*)d_in[7];
  const float* w_c2     = (const float*)d_in[8];
  const float* bc2      = (const float*)d_in[9];
  const float* g_cfe    = (const float*)d_in[10];
  const float* beta_cfe = (const float*)d_in[11];
  const float* w_c3     = (const float*)d_in[12];
  const float* bc3      = (const float*)d_in[13];
  const float* g2       = (const float*)d_in[14];
  const float* beta2    = (const float*)d_in[15];
  float* out = (float*)d_out;

  char* ws = (char*)d_ws;
  size_t off = 0;
  auto carve = [&](size_t bytes) -> char* {
    char* p = ws + off;
    off += (bytes + 255) & ~(size_t)255;
    return p;
  };
  _Float16* qkvT = (_Float16*)carve((size_t)Bq * C3 * PT * 2);              // 76 MB
  float*    pre1 = (float*)carve((size_t)TOT * 4);                          // 51 MB (reused as pre2)
  float*    x1   = (float*)carve((size_t)TOT * 4);                          // 51 MB
  _Float16* x1p  = (_Float16*)carve((size_t)Bq * CPAD * Pp * TPAD * 2);     // 52 MB halo buffer
  float*    y0   = (float*)carve((size_t)TOTY * 4);                         // 13 MB
  float*    yb   = (float*)carve((size_t)TOTY * 4);                         // 13 MB
  _Float16* whp  = (_Float16*)carve((size_t)WROWS * 256 * 2);
  _Float16* wcc  = (_Float16*)carve((size_t)FFD * KCP * 2);
  _Float16* relp = (_Float16*)carve((size_t)112 * 128 * 2);
  float* sc1 = (float*)carve(Cc * 4);  float* sh1 = (float*)carve(Cc * 4);
  float* sc2 = (float*)carve(FFD * 4); float* sh2 = (float*)carve(FFD * 4);
  float* sc3 = (float*)carve(Cc * 4);  float* sh3 = (float*)carve(Cc * 4);
  float* pre2 = pre1;  // pre1 dead after bn_apply_x1

  int prep_total = WROWS * 256 + FFD * KCP + 112 * 128;   // 230912
  prep_kernel<<<(prep_total + 255) / 256, 256, 0, stream>>>(w_qkv, w_c1, w_c2, rel_emb,
                                                            whp, wcc, relp);
  // zero the halo-padded im2col buffer (Bq*CPAD*Pp*TPAD halves / 8 per uint4)
  fill_zero<<<(Bq * CPAD * Pp * TPAD) / 8 / 256, 256, 0, stream>>>((uint4*)x1p);
  qkv_gemm<<<dim3((C3 + 63) / 64, (PT + 63) / 64, Bq), 256, 0, stream>>>(x, whp, b_qkv, qkvT);
  attn_kernel<<<dim3(Hh, Bq), 256, 102400, stream>>>(qkvT, relp, x, pre1);
  bn_stats<<<Cc, 256, 0, stream>>>(pre1, g1, beta1, sc1, sh1, Cc, PT, Bq);
  bn_apply_x1<<<TOT / 256, 256, 0, stream>>>(pre1, sc1, sh1, x1, x1p);
  cfe_gemm<<<NPT / 128, 256, 0, stream>>>(x1p, wcc, bc1, bc2, y0);
  bn_stats<<<FFD, 256, 0, stream>>>(y0, g_cfe, beta_cfe, sc2, sh2, FFD, PT, Bq);
  bn_elu<<<TOTY / 256, 256, 0, stream>>>(y0, sc2, sh2, yb);
  conv3_res<<<dim3(4, Bq), 256, 0, stream>>>(yb, w_c3, bc3, x1, pre2);
  bn_stats<<<Cc, 256, 0, stream>>>(pre2, g2, beta2, sc3, sh3, Cc, PT, Bq);
  bn_apply_out<<<TOT / 256, 256, 0, stream>>>(pre2, sc3, sh3, out);
}
